// OligoRNN_66949950210446
// MI455X (gfx1250) — matrix operand order repaired and checked by reading, not compiled, not run on hardware
//
#include <hip/hip_runtime.h>
#include <math.h>

typedef __attribute__((ext_vector_type(16))) _Float16 v16h;
typedef __attribute__((ext_vector_type(8)))  _Float16 v8h;
typedef __attribute__((ext_vector_type(8)))  float    v8f;

#define B_SZ  1024
#define T_SZ  128
#define IN_SZ 8
#define HENC  256
#define H_SZ  512
#define F_SZ  32
#define HF    (H_SZ + F_SZ)   // 544
#define BT    (B_SZ * T_SZ)   // 131072

// ---------- WMMA fragment loaders (per CDNA5 ISA 7.12.2 layouts) ----------

// A frag: rows m (lane&15), K tile k0..k0+31. lane<16: K=k0+[0..7],k0+[16..23];
// lane>=16: K=k0+[8..15],k0+[24..31].
static __device__ inline v16h load_a_frag(const _Float16* __restrict__ A, int lda, int k0) {
    const int lane = threadIdx.x & 31;
    const int half = lane >> 4;
    const int r    = lane & 15;
    const _Float16* p = A + (size_t)r * lda + k0 + 8 * half;
    v8h lo = *(const v8h*)(p);
    v8h hi = *(const v8h*)(p + 16);
    v16h a;
#pragma unroll
    for (int j = 0; j < 8; ++j) { a[j] = lo[j]; a[8 + j] = hi[j]; }
    return a;
}

// B frag for B[k][n] = W[n][k] with W stored [N,K] row-major.
// lane<16: col n=lane, K=k0+[0..7],k0+[16..23] ; lane>=16: col n=lane-16, K=k0+[8..15],k0+[24..31].
static __device__ inline v16h load_b_frag(const _Float16* __restrict__ W, int ldw, int n0, int k0) {
    const int lane = threadIdx.x & 31;
    const int half = lane >> 4;
    const int n    = lane & 15;
    const _Float16* p = W + (size_t)(n0 + n) * ldw + k0 + 8 * half;
    v8h lo = *(const v8h*)(p);
    v8h hi = *(const v8h*)(p + 16);
    v16h b;
#pragma unroll
    for (int j = 0; j < 8; ++j) { b[j] = lo[j]; b[8 + j] = hi[j]; }
    return b;
}

// ---------- weight conversion f32 -> f16 ----------
__global__ void cvt_f16(const float* __restrict__ src, _Float16* __restrict__ dst, int n) {
    int i = blockIdx.x * blockDim.x + threadIdx.x;
    if (i < n) dst[i] = (_Float16)src[i];
}

// ---------- encoder layer 1 (K=8, VALU) : out = relu(seq @ w1^T + b1), f16 ----------
__global__ void __launch_bounds__(256)
enc_layer1(const float* __restrict__ seq, const float* __restrict__ w1,
           const float* __restrict__ b1, _Float16* __restrict__ out) {
    const size_t row = blockIdx.x;
    const int n = threadIdx.x;                 // 0..255
    const float* x = seq + row * IN_SZ;
    const float* w = w1 + n * IN_SZ;
    float s = b1[n];
#pragma unroll
    for (int k = 0; k < IN_SZ; ++k) s += x[k] * w[k];
    out[row * HENC + n] = (_Float16)fmaxf(s, 0.f);
}

// ---------- wide WMMA GEMM (requires M%32==0, N%256==0 per grid coverage) ----------
// block: 8 waves = 2 m-tiles x 4 n-groups; each wave: 16 rows x 64 cols (4 acc).
// One A fragment feeds 4 WMMAs -> A traffic cut 4x vs narrow kernel.
__global__ void __launch_bounds__(256)
gemm_f16_wide(const _Float16* __restrict__ A, const _Float16* __restrict__ W,
              const float* __restrict__ bias, _Float16* __restrict__ C,
              int M, int N, int K, int relu) {
    const int wave = threadIdx.x >> 5;
    const int lane = threadIdx.x & 31;
    const int m0 = blockIdx.x * 32 + (wave >> 2) * 16;
    const int nb = blockIdx.y * 256 + (wave & 3) * 64;   // wave's 64-col base
    v8f acc[4] = {};
    const _Float16* Arow = A + (size_t)m0 * K;
    for (int k0 = 0; k0 < K; k0 += 32) {
        v16h a = load_a_frag(Arow, K, k0);
#pragma unroll
        for (int nt = 0; nt < 4; ++nt) {
            v16h b = load_b_frag(W, K, nb + nt * 16, k0);
            acc[nt] = __builtin_amdgcn_wmma_f32_16x16x32_f16(false, a, false, b, (short)0, acc[nt], false, false);
        }
    }
    const int half = lane >> 4;
    const int nc   = lane & 15;
#pragma unroll
    for (int nt = 0; nt < 4; ++nt) {
        const int n = nb + nt * 16 + nc;
        const float bn = bias[n];
#pragma unroll
        for (int r = 0; r < 8; ++r) {
            const int m = m0 + r + 8 * half;
            float v = acc[nt][r] + bn;
            if (relu) v = fmaxf(v, 0.f);
            C[(size_t)m * N + n] = (_Float16)v;
        }
    }
}

// ---------- narrow WMMA GEMM (handles N not divisible by 64; used for N=544) ----------
__global__ void __launch_bounds__(256)
gemm_f16(const _Float16* __restrict__ A, const _Float16* __restrict__ W,
         const float* __restrict__ bias, _Float16* __restrict__ C,
         int M, int N, int K, int relu) {
    const int wave = threadIdx.x >> 5;
    const int lane = threadIdx.x & 31;
    const int m0 = blockIdx.x * 32 + (wave >> 2) * 16;
    const int n0 = blockIdx.y * 64 + (wave & 3) * 16;
    if (m0 >= M || n0 >= N) return;
    v8f acc = {};
    const _Float16* Arow = A + (size_t)m0 * K;
    for (int k0 = 0; k0 < K; k0 += 32) {
        v16h a = load_a_frag(Arow, K, k0);
        v16h b = load_b_frag(W, K, n0, k0);
        acc = __builtin_amdgcn_wmma_f32_16x16x32_f16(false, a, false, b, (short)0, acc, false, false);
    }
    const int half = lane >> 4;
    const int n = n0 + (lane & 15);
    const float bn = bias[n];
#pragma unroll
    for (int r = 0; r < 8; ++r) {
        const int m = m0 + r + 8 * half;
        float v = acc[r] + bn;
        if (relu) v = fmaxf(v, 0.f);
        C[(size_t)m * N + n] = (_Float16)v;
    }
}

// ---------- RNN scan: block owns 16 batch rows; h lives in LDS across all T ----------
__global__ void __launch_bounds__(256)
rnn_scan(const _Float16* __restrict__ xw,    // [B,T,H] = enc @ W_ih^T + b_ih
         const _Float16* __restrict__ Whh,   // [H,H] f16
         const float* __restrict__ b_hh,
         const int* __restrict__ lengths,
         _Float16* __restrict__ hs) {        // [B,T,H] masked outputs
    __shared__ _Float16 hbuf[16][H_SZ + 8];  // +8 f16 pad -> LDS bank spread
    __shared__ int len_s[16];
    const int b0   = blockIdx.x * 16;
    const int wave = threadIdx.x >> 5;       // 0..7, each owns 64 output cols
    const int lane = threadIdx.x & 31;
    const int half = lane >> 4;
    const int nco  = lane & 15;

    for (int i = threadIdx.x; i < 16 * (H_SZ + 8); i += blockDim.x)
        (&hbuf[0][0])[i] = (_Float16)0.f;
    if (threadIdx.x < 16) len_s[threadIdx.x] = lengths[b0 + threadIdx.x];
    __syncthreads();

    for (int t = 0; t < T_SZ; ++t) {
        v8f acc[4] = {};
        for (int k0 = 0; k0 < H_SZ; k0 += 32) {
            v16h a = load_a_frag(&hbuf[0][0], H_SZ + 8, k0);
#pragma unroll
            for (int nt = 0; nt < 4; ++nt) {
                const int n0 = wave * 64 + nt * 16;
                v16h b = load_b_frag(Whh, H_SZ, n0, k0);
                acc[nt] = __builtin_amdgcn_wmma_f32_16x16x32_f16(false, a, false, b, (short)0, acc[nt], false, false);
            }
        }
        float hnew[4][8];
#pragma unroll
        for (int nt = 0; nt < 4; ++nt) {
            const int n = wave * 64 + nt * 16 + nco;
            const float bn = b_hh[n];
#pragma unroll
            for (int r = 0; r < 8; ++r) {
                const int ml = r + 8 * half;
                const int bg = b0 + ml;
                const size_t idx = ((size_t)bg * T_SZ + t) * H_SZ + n;
                float pre = acc[nt][r] + (float)xw[idx] + bn;
                float hn  = tanhf(pre);
                bool valid = t < len_s[ml];
                float hold = (float)hbuf[ml][n];
                hnew[nt][r] = valid ? hn : hold;
                hs[idx] = (_Float16)(valid ? hn : 0.f);
            }
        }
        __syncthreads();   // everyone done reading old h
#pragma unroll
        for (int nt = 0; nt < 4; ++nt) {
            const int n = wave * 64 + nt * 16 + nco;
#pragma unroll
            for (int r = 0; r < 8; ++r)
                hbuf[r + 8 * half][n] = (_Float16)hnew[nt][r];
        }
        __syncthreads();   // new h visible before next step's reads
    }
}

// ---------- shared-MLP layer2 fused with mask + max-pool over T ----------
// grid (B, H/128), 4 waves/block, each wave owns 32 cols (2 n-tiles, A reused).
__global__ void __launch_bounds__(128)
pool_mlp2(const _Float16* __restrict__ hid,  // [B,T,H] relu(hs @ sh_w1^T + b1)
          const _Float16* __restrict__ W2,   // [H,H]
          const float* __restrict__ b2,
          const int* __restrict__ lengths,
          _Float16* __restrict__ cat) {      // [B, HF]
    const int b    = blockIdx.x;
    const int wave = threadIdx.x >> 5;
    const int lane = threadIdx.x & 31;
    const int half = lane >> 4;
    const int nco  = lane & 15;
    const int n0   = blockIdx.y * 128 + wave * 32;   // wave's 32-col base
    const int len  = lengths[b];
    float rmax[2][8];
#pragma unroll
    for (int nt = 0; nt < 2; ++nt)
#pragma unroll
        for (int r = 0; r < 8; ++r) rmax[nt][r] = -3.402823466e38f;

    for (int tc = 0; tc < T_SZ; tc += 16) {
        v8f acc[2] = {};
        const _Float16* Abase = hid + ((size_t)b * T_SZ + tc) * H_SZ;
        for (int k0 = 0; k0 < H_SZ; k0 += 32) {
            v16h a = load_a_frag(Abase, H_SZ, k0);
#pragma unroll
            for (int nt = 0; nt < 2; ++nt) {
                v16h bf = load_b_frag(W2, H_SZ, n0 + nt * 16, k0);
                acc[nt] = __builtin_amdgcn_wmma_f32_16x16x32_f16(false, a, false, bf, (short)0, acc[nt], false, false);
            }
        }
#pragma unroll
        for (int nt = 0; nt < 2; ++nt) {
            const float bn = b2[n0 + nt * 16 + nco];
#pragma unroll
            for (int r = 0; r < 8; ++r) {
                const int t = tc + r + 8 * half;
                float v = acc[nt][r] + bn;
                v = (t < len) ? v : 0.f;      // masked positions contribute 0
                rmax[nt][r] = fmaxf(rmax[nt][r], v);
            }
        }
    }
#pragma unroll
    for (int nt = 0; nt < 2; ++nt) {
        float m = rmax[nt][0];
#pragma unroll
        for (int r = 1; r < 8; ++r) m = fmaxf(m, rmax[nt][r]);
        m = fmaxf(m, __shfl_xor(m, 16, 32));  // combine lane halves (rows)
        if (half == 0) cat[(size_t)b * HF + n0 + nt * 16 + nco] = (_Float16)m;
    }
}

// ---------- features MLP (F=32, VALU) -> cat[:, H:H+F] ----------
__global__ void __launch_bounds__(32)
feat_mlp(const float* __restrict__ feat, const float* __restrict__ w1,
         const float* __restrict__ b1, const float* __restrict__ w2,
         const float* __restrict__ b2, _Float16* __restrict__ cat) {
    __shared__ float x[F_SZ], h1[F_SZ];
    const int b = blockIdx.x, n = threadIdx.x;
    x[n] = feat[(size_t)b * F_SZ + n];
    __syncthreads();
    float s = b1[n];
#pragma unroll
    for (int k = 0; k < F_SZ; ++k) s += w1[n * F_SZ + k] * x[k];
    h1[n] = fmaxf(s, 0.f);
    __syncthreads();
    float o = b2[n];
#pragma unroll
    for (int k = 0; k < F_SZ; ++k) o += w2[n * F_SZ + k] * h1[k];
    cat[(size_t)b * HF + H_SZ + n] = (_Float16)o;
}

// ---------- final layer 2: dot(fhid[b,:], fn_w2) + fn_b2 ----------
__global__ void __launch_bounds__(32)
final_dot(const _Float16* __restrict__ fhid, const float* __restrict__ w2,
          const float* __restrict__ b2, float* __restrict__ out) {
    const int b = blockIdx.x, lane = threadIdx.x;
    float s = 0.f;
    for (int k = lane; k < HF; k += 32) s += (float)fhid[(size_t)b * HF + k] * w2[k];
#pragma unroll
    for (int off = 16; off; off >>= 1) s += __shfl_xor(s, off, 32);
    if (lane == 0) out[b] = s + b2[0];
}

// ---------------------------------------------------------------------------
extern "C" void kernel_launch(void* const* d_in, const int* in_sizes, int n_in,
                              void* d_out, int out_size, void* d_ws, size_t ws_size,
                              hipStream_t stream) {
    const float* seq      = (const float*)d_in[0];
    const float* features = (const float*)d_in[1];
    const int*   lengths  = (const int*)d_in[2];
    const float* enc_w1 = (const float*)d_in[3],  *enc_b1 = (const float*)d_in[4];
    const float* enc_w2 = (const float*)d_in[5],  *enc_b2 = (const float*)d_in[6];
    const float* W_ih   = (const float*)d_in[7],  *b_ih   = (const float*)d_in[8];
    const float* W_hh   = (const float*)d_in[9],  *b_hh   = (const float*)d_in[10];
    const float* sh_w1  = (const float*)d_in[11], *sh_b1  = (const float*)d_in[12];
    const float* sh_w2  = (const float*)d_in[13], *sh_b2  = (const float*)d_in[14];
    const float* ft_w1  = (const float*)d_in[15], *ft_b1  = (const float*)d_in[16];
    const float* ft_w2  = (const float*)d_in[17], *ft_b2  = (const float*)d_in[18];
    const float* fn_w1  = (const float*)d_in[19], *fn_b1  = (const float*)d_in[20];
    const float* fn_w2  = (const float*)d_in[21], *fn_b2  = (const float*)d_in[22];
    float* out = (float*)d_out;

    // workspace carve-up (256B aligned)
    char* ws = (char*)d_ws;
    size_t off = 0;
    auto alloc16 = [&](size_t nelem) {
        void* p = ws + off;
        off = (off + nelem * sizeof(_Float16) + 255) & ~(size_t)255;
        return (_Float16*)p;
    };
    _Float16* wEnc2 = alloc16((size_t)HENC * HENC);
    _Float16* wIh   = alloc16((size_t)H_SZ * HENC);
    _Float16* wHh   = alloc16((size_t)H_SZ * H_SZ);
    _Float16* wSh1  = alloc16((size_t)H_SZ * H_SZ);
    _Float16* wSh2  = alloc16((size_t)H_SZ * H_SZ);
    _Float16* wFn1  = alloc16((size_t)HF * HF);
    _Float16* hidB  = alloc16((size_t)BT * H_SZ);  // enc_h1 | enc_out, later reused as shared-MLP hidden
    _Float16* xw    = alloc16((size_t)BT * H_SZ);
    _Float16* hs    = alloc16((size_t)BT * H_SZ);
    _Float16* cat   = alloc16((size_t)B_SZ * HF);
    _Float16* fhid  = alloc16((size_t)B_SZ * HF);
    _Float16* enc_h1  = hidB;                         // [BT, HENC]
    _Float16* enc_out = hidB + (size_t)BT * HENC;     // [BT, HENC] (exactly second half)

    auto cvt = [&](const float* s, _Float16* d, int n) {
        cvt_f16<<<(n + 255) / 256, 256, 0, stream>>>(s, d, n);
    };
    cvt(enc_w2, wEnc2, HENC * HENC);
    cvt(W_ih,   wIh,   H_SZ * HENC);
    cvt(W_hh,   wHh,   H_SZ * H_SZ);
    cvt(sh_w1,  wSh1,  H_SZ * H_SZ);
    cvt(sh_w2,  wSh2,  H_SZ * H_SZ);
    cvt(fn_w1,  wFn1,  HF * HF);

    // 1) encoder layer1 (relu) -> enc_h1
    enc_layer1<<<BT, 256, 0, stream>>>(seq, enc_w1, enc_b1, enc_h1);
    // 2) encoder layer2 -> enc_out   (N=256 -> one 256-col group)
    gemm_f16_wide<<<dim3(BT / 32, HENC / 256), 256, 0, stream>>>(enc_h1, wEnc2, enc_b2, enc_out, BT, HENC, HENC, 0);
    // 3) xw = enc_out @ W_ih^T + b_ih   (N=512 -> two 256-col groups)
    gemm_f16_wide<<<dim3(BT / 32, H_SZ / 256), 256, 0, stream>>>(enc_out, wIh, b_ih, xw, BT, H_SZ, HENC, 0);
    // 4) RNN scan -> hs (masked)
    rnn_scan<<<B_SZ / 16, 256, 0, stream>>>(xw, wHh, b_hh, lengths, hs);
    // 5) shared layer1 (relu) -> hid (reuses encoder buffers)
    gemm_f16_wide<<<dim3(BT / 32, H_SZ / 256), 256, 0, stream>>>(hs, wSh1, sh_b1, hidB, BT, H_SZ, H_SZ, 1);
    // 6) shared layer2 + mask + max-pool -> cat[:, 0:H]
    pool_mlp2<<<dim3(B_SZ, H_SZ / 128), 128, 0, stream>>>(hidB, wSh2, sh_b2, lengths, cat);
    // 7) features MLP -> cat[:, H:H+F]
    feat_mlp<<<B_SZ, 32, 0, stream>>>(features, ft_w1, ft_b1, ft_w2, ft_b2, cat);
    // 8) final layer1 (relu) -> fhid  (N=544 not /64: narrow kernel with guards)
    gemm_f16<<<dim3(B_SZ / 32, (HF + 63) / 64), 256, 0, stream>>>(cat, wFn1, fn_b1, fhid, B_SZ, HF, HF, 1);
    // 9) final dot -> out
    final_dot<<<B_SZ, 32, 0, stream>>>(fhid, fn_w2, fn_b2, out);
}
